// FusionRestormerBlock_4561255268509
// MI455X (gfx1250) — compile-verified, matrix operand order checked by reading, not run on hardware
//
#include <hip/hip_runtime.h>
#include <hip/hip_bf16.h>
#include <math.h>

typedef __attribute__((ext_vector_type(16))) __bf16 v16bf;
typedef __attribute__((ext_vector_type(8)))  __bf16 v8bf;
typedef __attribute__((ext_vector_type(8)))  float  v8f;

#define DEV static __device__ __forceinline__

namespace {

constexpr float kScale = 0.2041241452319315f;   // 1/sqrt(24)
constexpr float kEps   = 1e-6f;

// bf16-weight workspace layout (element offsets into d_ws)
constexpr int OFF_QKV = 0;                  // (576,192)
constexpr int OFF_AP  = OFF_QKV + 576*192;  // (192,192)
constexpr int OFF_Q   = OFF_AP  + 192*192;  // (192,192)
constexpr int OFF_KV  = OFF_Q   + 192*192;  // (384,192)
constexpr int OFF_CP  = OFF_KV  + 384*192;  // (192,192)
constexpr int OFF_F1  = OFF_CP  + 192*192;  // (768,192)
constexpr int OFF_F2  = OFF_F1  + 768*192;  // (192,768)
constexpr int W_TOTAL = OFF_F2  + 192*768;  // 589824 elements

// dynamic-LDS byte offsets (one 64-token window per workgroup)
constexpr int SM_X0   = 0;        // float [64][192]   48 KB  (x / x1 / x2 residual)
constexpr int SM_XN   = 49152;    // bf16  [64][192]   24 KB  (cln out / O merge / SAR)
constexpr int SM_Q    = 73728;    // bf16  [8][64][32] 32 KB  (head-padded Q)
constexpr int SM_K    = 106496;   // bf16  [8][64][32] 32 KB
constexpr int SM_VT   = 139264;   // bf16  [8][32][64] 32 KB  (V transposed per head)
constexpr int SM_S    = 172032;   // float [8][64][64] 128 KB (per-wave scores)
constexpr int SM_RS   = 303104;   // float [8][64]     2 KB   (1/rowsum)
constexpr int SM_RELB = 305152;   // float [225][8]    7.2 KB
constexpr int SM_RED  = 312352;   // float [512]       2 KB
constexpr int SM_MU   = 314400;   // float [64]
constexpr int SM_RSTD = 314656;   // float [64]
constexpr int SMEM_BYTES = 314912;

DEV v8f vzero() {
  v8f a;
  #pragma unroll
  for (int i = 0; i < 8; ++i) a[i] = 0.f;
  return a;
}

// 16-bit A-matrix 16x32 lane layout: lane m=lane&15, half=lane>>4;
// elements 0..7 -> K = 8*half + 0..7 ; elements 8..15 -> K = 16 + 8*half + 0..7
DEV v16bf ldA_bf(const __bf16* p) {
  const int half = (threadIdx.x & 31) >> 4;
  v8bf lo = *(const v8bf*)(p + 8 * half);
  v8bf hi = *(const v8bf*)(p + 16 + 8 * half);
  v16bf r;
  #pragma unroll
  for (int i = 0; i < 8; ++i) { r[i] = lo[i]; r[i + 8] = hi[i]; }
  return r;
}

DEV v16bf ldA_f32(const float* p) {   // same layout, convert f32 -> bf16 on the fly
  const int half = (threadIdx.x & 31) >> 4;
  v16bf r;
  #pragma unroll
  for (int i = 0; i < 8; ++i) {
    r[i]     = (__bf16)p[8 * half + i];
    r[i + 8] = (__bf16)p[16 + 8 * half + i];
  }
  return r;
}

DEV v8f wmma_bf(v16bf a, v16bf b, v8f c) {
  return __builtin_amdgcn_wmma_f32_16x16x32_bf16(false, a, false, b, (short)0, c,
                                                 false, false);
}

// D = A(64xK) x B^T(NxK row-major) tile; A rows from mBase, B rows from nBase
DEV v8f gemm_tile(const __bf16* A, int mBase, int sA,
                  const __bf16* B, int nBase, int sB, int ksteps) {
  const int lane = threadIdx.x & 31;
  const __bf16* ap = A + (mBase + (lane & 15)) * sA;
  const __bf16* bp = B + (nBase + (lane & 15)) * sB;
  v8f acc = vzero();
  for (int kb = 0; kb < ksteps; ++kb)
    acc = wmma_bf(ldA_bf(ap + kb * 32), ldA_bf(bp + kb * 32), acc);
  return acc;
}

// conditional LayerNorm over channels: out = gamma*(lnw*xhat+lnb)+beta, bf16
DEV void cln(const float* Xf, __bf16* out,
             const float* lnw, const float* lnb,
             const float* gb, const float* bb,
             float* red, float* Mu, float* Rstd) {
  const int tid = threadIdx.x, l = tid & 63, part = tid >> 6;
  const float* row = Xf + l * 192 + part * 48;
  float s1 = 0.f, s2 = 0.f;
  for (int i = 0; i < 48; ++i) { float v = row[i]; s1 += v; s2 += v * v; }
  red[part * 64 + l] = s1; red[256 + part * 64 + l] = s2;
  __syncthreads();
  if (tid < 64) {
    float a = 0.f, c2 = 0.f;
    for (int p = 0; p < 4; ++p) { a += red[p * 64 + tid]; c2 += red[256 + p * 64 + tid]; }
    const float mu = a * (1.f / 192.f);
    const float var = c2 * (1.f / 192.f) - mu * mu;
    Mu[tid] = mu; Rstd[tid] = rsqrtf(var + kEps);
  }
  __syncthreads();
  const float mu = Mu[l], rs = Rstd[l];
  for (int i = 0; i < 48; ++i) {
    const int c = part * 48 + i;
    const float xn = (Xf[l * 192 + c] - mu) * rs * lnw[c] + lnb[c];
    out[l * 192 + c] = (__bf16)(gb[c] * xn + bb[c]);
  }
  __syncthreads();
}

// one head per wave: scores -> softmax -> O, writes merged heads into Obf[64][192]
DEV void attention(const __bf16* Qb, const __bf16* Kb, const __bf16* Vt,
                   float* S, float* RSs, const float* relb, __bf16* Obf,
                   bool useBias) {
  const int tid = threadIdx.x, h = tid >> 5, lane = tid & 31;
  const int nloc = lane & 15, half = lane >> 4;
  float* Sw = S + h * 4096;
  const __bf16* Qh = Qb + h * 2048;
  const __bf16* Kh = Kb + h * 2048;
  const __bf16* Vh = Vt + h * 2048;

  // scores: S = (Q K^T)*scale (+ rel bias)
  for (int mt = 0; mt < 4; ++mt)
    for (int nt = 0; nt < 4; ++nt) {
      v8f acc = wmma_bf(ldA_bf(Qh + (mt * 16 + nloc) * 32),
                        ldA_bf(Kh + (nt * 16 + nloc) * 32), vzero());
      const int nG = nt * 16 + nloc;
      #pragma unroll
      for (int r = 0; r < 8; ++r) {
        const int mG = mt * 16 + r + 8 * half;
        float sv = acc[r] * kScale;
        if (useBias) {
          const int dr = (mG >> 3) - (nG >> 3) + 7;
          const int dc = (mG & 7) - (nG & 7) + 7;
          sv += relb[(dr * 15 + dc) * 8 + h];
        }
        Sw[mG * 64 + nG] = sv;
      }
    }
  __syncthreads();

  // softmax rows (2 rows per lane), keep unnormalized exp, store 1/sum
  for (int rr = 0; rr < 2; ++rr) {
    const int row = lane * 2 + rr;
    float* sp = Sw + row * 64;
    float mx = -3.0e38f;
    for (int j = 0; j < 64; ++j) mx = fmaxf(mx, sp[j]);
    float sum = 0.f;
    for (int j = 0; j < 64; ++j) { float e = __expf(sp[j] - mx); sp[j] = e; sum += e; }
    RSs[h * 64 + row] = 1.f / sum;
  }
  __syncthreads();

  // O = softmax(S) @ V  (V stored transposed, HD padded to 32 with zeros)
  for (int mt = 0; mt < 4; ++mt)
    for (int nt = 0; nt < 2; ++nt) {
      v8f acc = vzero();
      for (int kb = 0; kb < 2; ++kb)
        acc = wmma_bf(ldA_f32(Sw + (mt * 16 + nloc) * 64 + kb * 32),
                      ldA_bf(Vh + (nt * 16 + nloc) * 64 + kb * 32), acc);
      const int d = nt * 16 + nloc;
      if (d < 24) {
        #pragma unroll
        for (int r = 0; r < 8; ++r) {
          const int m = mt * 16 + r + 8 * half;
          Obf[m * 192 + h * 24 + d] = (__bf16)(acc[r] * RSs[h * 64 + m]);
        }
      }
    }
  __syncthreads();
}

__global__ void convert_weights_k(const float* qkv, const float* ap, const float* qw,
                                  const float* kv, const float* cp, const float* f1,
                                  const float* f2, __bf16* ws) {
  int i = blockIdx.x * blockDim.x + threadIdx.x;
  if (i >= W_TOTAL) return;
  const float* src; int base;
  if      (i < OFF_AP) { src = qkv; base = OFF_QKV; }
  else if (i < OFF_Q ) { src = ap;  base = OFF_AP;  }
  else if (i < OFF_KV) { src = qw;  base = OFF_Q;   }
  else if (i < OFF_CP) { src = kv;  base = OFF_KV;  }
  else if (i < OFF_F1) { src = cp;  base = OFF_CP;  }
  else if (i < OFF_F2) { src = f1;  base = OFF_F1;  }
  else                 { src = f2;  base = OFF_F2;  }
  ws[i] = (__bf16)src[i - base];
}

__global__ __launch_bounds__(256, 1)
void fused_window_block(const float* __restrict__ x_rgb,
                        const float* __restrict__ x_sar,
                        const float* __restrict__ gamma,
                        const float* __restrict__ beta,
                        const float* __restrict__ ln1w, const float* __restrict__ ln1b,
                        const float* __restrict__ ln2w, const float* __restrict__ ln2b,
                        const float* __restrict__ qkv_b,
                        const float* __restrict__ ap_b,
                        const float* __restrict__ rel_bias,
                        const float* __restrict__ cp_b,
                        const __bf16* __restrict__ ws,
                        float* __restrict__ out) {
  extern __shared__ char smem[];
  float*  X0   = (float*)(smem + SM_X0);
  __bf16* XN   = (__bf16*)(smem + SM_XN);
  __bf16* Qb   = (__bf16*)(smem + SM_Q);
  __bf16* Kb   = (__bf16*)(smem + SM_K);
  __bf16* Vt   = (__bf16*)(smem + SM_VT);
  float*  S    = (float*)(smem + SM_S);
  float*  RSs  = (float*)(smem + SM_RS);
  float*  relb = (float*)(smem + SM_RELB);
  float*  red  = (float*)(smem + SM_RED);
  float*  Mu   = (float*)(smem + SM_MU);
  float*  Rstd = (float*)(smem + SM_RSTD);
  __bf16* Hbf  = Qb;   // FFN hidden [64][768] aliases Q/K/Vt (96 KB)

  const int tid = threadIdx.x, w = tid >> 5, lane = tid & 31;
  const int nloc = lane & 15, half = lane >> 4;
  const int l = tid & 63, part = tid >> 6;

  // window decode
  const int n = blockIdx.x;
  const int b = n >> 8, rem = n & 255, wh = rem >> 4, wwv = rem & 15;
  const int hh = wh * 8 + (l >> 3), wwp = wwv * 8 + (l & 7);
  const size_t pix = (size_t)hh * 128 + wwp;

  const float* gb = gamma + b * 192;
  const float* bb = beta + b * 192;
  const __bf16* Wqkv = ws + OFF_QKV;
  const __bf16* Wap  = ws + OFF_AP;
  const __bf16* Wq   = ws + OFF_Q;
  const __bf16* Wkv  = ws + OFF_KV;
  const __bf16* Wcp  = ws + OFF_CP;
  const __bf16* Wf1  = ws + OFF_F1;
  const __bf16* Wf2  = ws + OFF_F2;

  // init: zero padded Q/K/Vt region, stage rel_bias, load x_rgb window
  for (int i = tid; i < 24576; i += 256) ((unsigned*)(smem + SM_Q))[i] = 0u;
  for (int i = tid; i < 225 * 8; i += 256) relb[i] = rel_bias[i];
  {
    const float* src = x_rgb + (size_t)b * 192 * 16384 + pix;
    for (int i = 0; i < 48; ++i) {
      const int c = part * 48 + i;
      X0[l * 192 + c] = src[(size_t)c * 16384];
    }
  }
  __syncthreads();

  // ---------------- self attention ----------------
  cln(X0, XN, ln1w, ln1b, gb, bb, red, Mu, Rstd);

  // QKV = cln(x) @ qkv_w^T + qkv_b, scatter into padded head layout
  for (int job = w; job < 144; job += 8) {
    const int mt = job & 3, nt = job >> 2;
    v8f acc = gemm_tile(XN, mt * 16, 192, Wqkv, nt * 16, 192, 6);
    const int col = nt * 16 + nloc;
    const int which = col / 192, c = col % 192, head = c / 24, d = c % 24;
    const float qb = qkv_b[col];
    #pragma unroll
    for (int r = 0; r < 8; ++r) {
      const int m = mt * 16 + r + 8 * half;
      const float v = acc[r] + qb;
      if (which == 0)      Qb[head * 2048 + m * 32 + d] = (__bf16)v;
      else if (which == 1) Kb[head * 2048 + m * 32 + d] = (__bf16)v;
      else                 Vt[head * 2048 + d * 64 + m] = (__bf16)v;
    }
  }
  __syncthreads();

  attention(Qb, Kb, Vt, S, RSs, relb, XN, /*useBias=*/true);

  // x1 = x + O @ attn_proj_w^T + b   (in place into X0)
  for (int job = w; job < 48; job += 8) {
    const int mt = job & 3, nt = job >> 2;
    v8f acc = gemm_tile(XN, mt * 16, 192, Wap, nt * 16, 192, 6);
    const int col = nt * 16 + nloc;
    const float pb = ap_b[col];
    #pragma unroll
    for (int r = 0; r < 8; ++r) {
      const int m = mt * 16 + r + 8 * half;
      X0[m * 192 + col] += acc[r] + pb;
    }
  }
  __syncthreads();

  // ---------------- cross attention ----------------
  cln(X0, XN, ln1w, ln1b, gb, bb, red, Mu, Rstd);

  // Q = cln(x1) @ q_w^T   (no bias)
  for (int job = w; job < 48; job += 8) {
    const int mt = job & 3, nt = job >> 2;
    v8f acc = gemm_tile(XN, mt * 16, 192, Wq, nt * 16, 192, 6);
    const int col = nt * 16 + nloc, head = col / 24, d = col % 24;
    #pragma unroll
    for (int r = 0; r < 8; ++r) {
      const int m = mt * 16 + r + 8 * half;
      Qb[head * 2048 + m * 32 + d] = (__bf16)acc[r];
    }
  }
  __syncthreads();

  // load SAR window (no LN) as bf16 into XN
  {
    const float* src = x_sar + (size_t)b * 192 * 16384 + pix;
    for (int i = 0; i < 48; ++i) {
      const int c = part * 48 + i;
      XN[l * 192 + c] = (__bf16)src[(size_t)c * 16384];
    }
  }
  __syncthreads();

  // K,V = sar @ kv_w^T   (no bias)
  for (int job = w; job < 96; job += 8) {
    const int mt = job & 3, nt = job >> 2;
    v8f acc = gemm_tile(XN, mt * 16, 192, Wkv, nt * 16, 192, 6);
    const int col = nt * 16 + nloc;
    const int which = col / 192, c = col % 192, head = c / 24, d = c % 24;
    #pragma unroll
    for (int r = 0; r < 8; ++r) {
      const int m = mt * 16 + r + 8 * half;
      if (which == 0) Kb[head * 2048 + m * 32 + d] = (__bf16)acc[r];
      else            Vt[head * 2048 + d * 64 + m] = (__bf16)acc[r];
    }
  }
  __syncthreads();

  attention(Qb, Kb, Vt, S, RSs, relb, XN, /*useBias=*/false);

  // x2 = x1 + O @ cross_proj_w^T + b
  for (int job = w; job < 48; job += 8) {
    const int mt = job & 3, nt = job >> 2;
    v8f acc = gemm_tile(XN, mt * 16, 192, Wcp, nt * 16, 192, 6);
    const int col = nt * 16 + nloc;
    const float pb = cp_b[col];
    #pragma unroll
    for (int r = 0; r < 8; ++r) {
      const int m = mt * 16 + r + 8 * half;
      X0[m * 192 + col] += acc[r] + pb;
    }
  }
  __syncthreads();

  // ---------------- FFN ----------------
  cln(X0, XN, ln2w, ln2b, gb, bb, red, Mu, Rstd);

  // hidden = GELU(cln(x2) @ ffn_w1^T), bf16 into Hbf (aliases Q/K/Vt)
  for (int job = w; job < 192; job += 8) {
    const int mt = job & 3, nt = job >> 2;
    v8f acc = gemm_tile(XN, mt * 16, 192, Wf1, nt * 16, 192, 6);
    const int col = nt * 16 + nloc;
    #pragma unroll
    for (int r = 0; r < 8; ++r) {
      const int m = mt * 16 + r + 8 * half;
      const float x = acc[r];
      const float g = 0.5f * x * (1.f + erff(x * 0.70710678118654752f));
      Hbf[m * 768 + col] = (__bf16)g;
    }
  }
  __syncthreads();

  // out = x2 + hidden @ ffn_w2^T   (into X0)
  for (int job = w; job < 48; job += 8) {
    const int mt = job & 3, nt = job >> 2;
    v8f acc = gemm_tile(Hbf, mt * 16, 768, Wf2, nt * 16, 768, 24);
    const int col = nt * 16 + nloc;
    #pragma unroll
    for (int r = 0; r < 8; ++r) {
      const int m = mt * 16 + r + 8 * half;
      X0[m * 192 + col] += acc[r];
    }
  }
  __syncthreads();

  // coalesced writeout
  {
    float* dst = out + (size_t)b * 192 * 16384 + pix;
    for (int i = 0; i < 48; ++i) {
      const int c = part * 48 + i;
      dst[(size_t)c * 16384] = X0[l * 192 + c];
    }
  }
}

} // namespace

extern "C" void kernel_launch(void* const* d_in, const int* in_sizes, int n_in,
                              void* d_out, int out_size, void* d_ws, size_t ws_size,
                              hipStream_t stream) {
  (void)in_sizes; (void)n_in; (void)out_size; (void)ws_size;
  const float* x_rgb = (const float*)d_in[0];
  const float* x_sar = (const float*)d_in[1];
  const float* gamma = (const float*)d_in[2];
  const float* beta  = (const float*)d_in[3];
  const float* ln1w  = (const float*)d_in[4];
  const float* ln1b  = (const float*)d_in[5];
  const float* ln2w  = (const float*)d_in[6];
  const float* ln2b  = (const float*)d_in[7];
  const float* qkv_w = (const float*)d_in[8];
  const float* qkv_b = (const float*)d_in[9];
  const float* ap_w  = (const float*)d_in[10];
  const float* ap_b  = (const float*)d_in[11];
  const float* relb  = (const float*)d_in[12];
  const float* q_w   = (const float*)d_in[13];
  const float* kv_w  = (const float*)d_in[14];
  const float* cp_w  = (const float*)d_in[15];
  const float* cp_b  = (const float*)d_in[16];
  const float* f1_w  = (const float*)d_in[17];
  const float* f2_w  = (const float*)d_in[18];
  float* out = (float*)d_out;
  __bf16* ws = (__bf16*)d_ws;

  (void)hipFuncSetAttribute((const void*)fused_window_block,
                            hipFuncAttributeMaxDynamicSharedMemorySize, SMEM_BYTES);

  convert_weights_k<<<(W_TOTAL + 255) / 256, 256, 0, stream>>>(
      qkv_w, ap_w, q_w, kv_w, cp_w, f1_w, f2_w, ws);

  fused_window_block<<<2048, 256, SMEM_BYTES, stream>>>(
      x_rgb, x_sar, gamma, beta, ln1w, ln1b, ln2w, ln2b,
      qkv_b, ap_b, relb, cp_b, ws, out);
}
